// ChebNet_41120016892643
// MI455X (gfx1250) — compile-verified, matrix operand order checked by reading, not run on hardware
//
#include <hip/hip_runtime.h>
#include <hip/hip_bf16.h>

typedef __attribute__((ext_vector_type(16))) __bf16         v16bf;
typedef __attribute__((ext_vector_type(8)))  float          v8f;
typedef __attribute__((ext_vector_type(8)))  unsigned int   v8u;
typedef __attribute__((ext_vector_type(4)))  float          v4f;
typedef __attribute__((ext_vector_type(2)))  float          v2f;
typedef __attribute__((ext_vector_type(4)))  unsigned short v4us;
typedef __attribute__((ext_vector_type(2)))  unsigned int   v2u;

__device__ __forceinline__ unsigned short f32_to_bf16(float f) {
    unsigned int u = __builtin_bit_cast(unsigned int, f);
    unsigned int r = u + 0x7FFFu + ((u >> 16) & 1u);   // round-to-nearest-even
    return (unsigned short)(r >> 16);
}

__device__ __forceinline__ unsigned int pack_bf16x2(float lo, float hi) {
    return (unsigned int)f32_to_bf16(lo) | ((unsigned int)f32_to_bf16(hi) << 16);
}

// ---------------------------------------------------------------------------
// row_ptr[r] = lower_bound(edge_row, r); edge_row is sorted.
// ---------------------------------------------------------------------------
__global__ void build_row_ptr_kernel(const int* __restrict__ edge_row, int n_edges,
                                     int* __restrict__ row_ptr, int n_rows) {
    int r = blockIdx.x * blockDim.x + threadIdx.x;
    if (r > n_rows) return;
    int lo = 0, hi = n_edges;
    while (lo < hi) {
        int mid = (lo + hi) >> 1;
        if (edge_row[mid] < r) lo = mid + 1; else hi = mid;
    }
    row_ptr[r] = lo;
}

// ---------------------------------------------------------------------------
// Tiled GEMM on v_wmma_f32_16x16x32_bf16.
//   C[M,Nc] = act(A[M,K] @ B[K,Nc] + bias)
// Block: 128 threads = 4 waves; block tile 64x64, K-step 32.
// Wave w owns rows [16w,16w+16) x 64 cols = four 16x16 WMMA tiles.
// Staging is phase-split (all global loads issued, then convert+LDS store)
// so the b128 loads overlap; only the boundary M-block is scalar/guarded.
// ---------------------------------------------------------------------------
template <bool A_BF16, bool RELU, bool OUT_BF16>
__global__ __launch_bounds__(128)
void gemm_wmma_kernel(const void* __restrict__ Aptr, const float* __restrict__ B,
                      const float* __restrict__ bias, void* __restrict__ Cptr,
                      int M, int K, int Nc) {
    constexpr int BM = 64, BN = 64, BK = 32;
    constexpr int LDA = 40;                       // bf16 elems per LDS row (8B-aligned rows)
    __shared__ unsigned short As[BM * LDA];
    __shared__ unsigned short Bs[BN * LDA];

    const int tid  = threadIdx.x;
    const int wave = tid >> 5;
    const int lane = tid & 31;
    const int half = lane >> 4;                   // 0: lanes 0-15, 1: lanes 16-31
    const int lr   = lane & 15;

    const int m0 = blockIdx.x * BM;
    const int n0 = blockIdx.y * BN;
    const bool interior = (m0 + BM <= M);

    // per-thread staging coordinates
    const int ar = tid >> 3;                      // A row within tile (rep adds 16)
    const int aq = tid & 7;                       // A quad within row
    const int bk = tid >> 4;                      // B k within tile (rep adds 8)
    const int bn = (tid & 15) * 4;                // B col quad start

    v8f acc[4];
#pragma unroll
    for (int i = 0; i < 4; ++i) acc[i] = (v8f)0.0f;

    for (int k0 = 0; k0 < K; k0 += BK) {
        // ---- stage B tile transposed: Bs[n][k] = B[k0+k][n0+n] ----
        v4f bq[4];
#pragma unroll
        for (int rep = 0; rep < 4; ++rep)
            bq[rep] = *(const v4f*)(B + (size_t)(k0 + bk + rep * 8) * Nc + n0 + bn);

        // ---- stage A tile (64 x 32) as bf16 ----
        if (interior) {
            v2u ap[4];
            if constexpr (A_BF16) {
                v4us aq4[4];
#pragma unroll
                for (int rep = 0; rep < 4; ++rep)
                    aq4[rep] = *(const v4us*)((const unsigned short*)Aptr +
                                              (size_t)(m0 + ar + rep * 16) * K + k0 + aq * 4);
#pragma unroll
                for (int rep = 0; rep < 4; ++rep) {
                    ap[rep].x = (unsigned int)aq4[rep].x | ((unsigned int)aq4[rep].y << 16);
                    ap[rep].y = (unsigned int)aq4[rep].z | ((unsigned int)aq4[rep].w << 16);
                }
            } else {
                v4f af[4];
#pragma unroll
                for (int rep = 0; rep < 4; ++rep)
                    af[rep] = *(const v4f*)((const float*)Aptr +
                                            (size_t)(m0 + ar + rep * 16) * K + k0 + aq * 4);
#pragma unroll
                for (int rep = 0; rep < 4; ++rep) {
                    ap[rep].x = pack_bf16x2(af[rep].x, af[rep].y);
                    ap[rep].y = pack_bf16x2(af[rep].z, af[rep].w);
                }
            }
#pragma unroll
            for (int rep = 0; rep < 4; ++rep)
                *(v2u*)(As + (ar + rep * 16) * LDA + aq * 4) = ap[rep];

            // speculative prefetch of next K-chunk of A (global_prefetch_b8)
            if (k0 + BK < K) {
                size_t nofs = (size_t)(m0 + ar) * K + (k0 + BK) + aq * 4;
                __builtin_prefetch((const char*)Aptr + nofs * (A_BF16 ? 2 : 4), 0, 0);
            }
        } else {
            // boundary M-block: guarded scalar path, zero-fill out-of-range rows
#pragma unroll
            for (int rep = 0; rep < 16; ++rep) {
                int flat = rep * 128 + tid;
                int r = flat >> 5;
                int c = flat & 31;
                int gr = m0 + r;
                unsigned short v = 0;
                if (gr < M) {
                    if constexpr (A_BF16)
                        v = ((const unsigned short*)Aptr)[(size_t)gr * K + k0 + c];
                    else
                        v = f32_to_bf16(((const float*)Aptr)[(size_t)gr * K + k0 + c]);
                }
                As[r * LDA + c] = v;
            }
        }

        // ---- convert + store B tile ----
#pragma unroll
        for (int rep = 0; rep < 4; ++rep) {
            int k = bk + rep * 8;
            Bs[(bn + 0) * LDA + k] = f32_to_bf16(bq[rep].x);
            Bs[(bn + 1) * LDA + k] = f32_to_bf16(bq[rep].y);
            Bs[(bn + 2) * LDA + k] = f32_to_bf16(bq[rep].z);
            Bs[(bn + 3) * LDA + k] = f32_to_bf16(bq[rep].w);
        }
        __syncthreads();

        // ---- A fragment: 16-bit A layout, lane(half,lr), vgpr j -> K pair ----
        const unsigned int* Arow = (const unsigned int*)(As + (wave * 16 + lr) * LDA);
        v8u au;
#pragma unroll
        for (int j = 0; j < 8; ++j) {
            int kk = (j < 4) ? (2 * j + 8 * half) : (2 * j + 8 + 8 * half);
            au[j] = Arow[kk >> 1];
        }
        v16bf afrag = __builtin_bit_cast(v16bf, au);

#pragma unroll
        for (int nt = 0; nt < 4; ++nt) {
            const unsigned int* Brow = (const unsigned int*)(Bs + (nt * 16 + lr) * LDA);
            v8u bu;
#pragma unroll
            for (int j = 0; j < 8; ++j) {
                int kk = (j < 4) ? (2 * j + 8 * half) : (2 * j + 8 + 8 * half);
                bu[j] = Brow[kk >> 1];
            }
            v16bf bfrag = __builtin_bit_cast(v16bf, bu);
            acc[nt] = __builtin_amdgcn_wmma_f32_16x16x32_bf16(
                false, afrag, false, bfrag, (short)0, acc[nt], false, false);
        }
        __syncthreads();
    }

    // ---- epilogue: C layout M = j + 8*half, N = lr ----
#pragma unroll
    for (int nt = 0; nt < 4; ++nt) {
        int col = n0 + nt * 16 + lr;
        float bv = bias[col];
#pragma unroll
        for (int j = 0; j < 8; ++j) {
            int row = m0 + wave * 16 + j + 8 * half;
            if (row < M) {
                float v = acc[nt][j] + bv;
                if constexpr (RELU) v = v > 0.0f ? v : 0.0f;
                if constexpr (OUT_BF16)
                    ((unsigned short*)Cptr)[(size_t)row * Nc + col] = f32_to_bf16(v);
                else
                    ((float*)Cptr)[(size_t)row * Nc + col] = v;
            }
        }
    }
}

// ---------------------------------------------------------------------------
// z = gamma[0] * h   (full overwrite of d_out, float4 vectorized)
// ---------------------------------------------------------------------------
__global__ void zinit_kernel(const float* __restrict__ h, const float* __restrict__ gamma,
                             float* __restrict__ z, int nquads) {
    int i = blockIdx.x * blockDim.x + threadIdx.x;
    if (i < nquads) {
        v4f t = *(const v4f*)(h + (size_t)i * 4);
        *(v4f*)(z + (size_t)i * 4) = gamma[0] * t;
    }
}

// ---------------------------------------------------------------------------
// Fused SpMM + Chebyshev step. ONE WAVE (32 threads) per row; lane c owns the
// contiguous channel pair (2c, 2c+1) so every gather is a single b64 load.
// Edge (col,val) pairs staged through LDS in 32-edge chunks (mean degree=32);
// inner loop unrolled x4 with 4 independent float2 accumulator chains.
// The [N,64] f32 gather source (25.6 MB) lives in the 192 MB L2.
//   s   = sum_e vals[e] * v[col[e]][c]
//   t   = alpha*s - (use_prev ? prev[r][c] : 0)
//   z  += gamma[gidx] * t ;  out[r][c] = t
// ---------------------------------------------------------------------------
__global__ __launch_bounds__(32)
void spmm_cheb_kernel(const float* __restrict__ v, const float* __restrict__ prevb,
                      float* __restrict__ outb, float* __restrict__ z,
                      const int* __restrict__ row_ptr, const int* __restrict__ col,
                      const float* __restrict__ vals, const float* __restrict__ gamma,
                      int gidx, float alpha, int use_prev) {
    __shared__ int   sc[32];
    __shared__ float sv[32];
    const int r = blockIdx.x;
    const int c = threadIdx.x;                     // 0..31, channel pair (2c, 2c+1)
    const int e0 = row_ptr[r];
    const int e1 = row_ptr[r + 1];

    v2f s0 = (v2f)0.0f, s1 = (v2f)0.0f, s2 = (v2f)0.0f, s3 = (v2f)0.0f;
    for (int base = e0; base < e1; base += 32) {
        int cnt = e1 - base;
        if (cnt > 32) cnt = 32;
        __syncthreads();                           // single-wave WG: near-free
        if (c < cnt) {
            sc[c] = col[base + c];
            sv[c] = vals[base + c];
        }
        __syncthreads();
        __builtin_prefetch(col + base + 32, 0, 0);   // speculative, OOB-safe
        __builtin_prefetch(vals + base + 32, 0, 0);
        int j = 0;
        for (; j + 4 <= cnt; j += 4) {
            v2f g0 = *(const v2f*)(v + (size_t)sc[j + 0] * 64 + 2 * c);
            v2f g1 = *(const v2f*)(v + (size_t)sc[j + 1] * 64 + 2 * c);
            v2f g2 = *(const v2f*)(v + (size_t)sc[j + 2] * 64 + 2 * c);
            v2f g3 = *(const v2f*)(v + (size_t)sc[j + 3] * 64 + 2 * c);
            s0 += sv[j + 0] * g0;
            s1 += sv[j + 1] * g1;
            s2 += sv[j + 2] * g2;
            s3 += sv[j + 3] * g3;
        }
        for (; j < cnt; ++j)
            s0 += sv[j] * *(const v2f*)(v + (size_t)sc[j] * 64 + 2 * c);
    }

    v2f s = (s0 + s1) + (s2 + s3);
    v2f t = alpha * s;
    size_t o = (size_t)r * 64 + 2 * c;
    if (use_prev) t -= *(const v2f*)(prevb + o);
    float g = gamma[gidx];
    v2f zt = *(const v2f*)(z + o) + g * t;
    *(v2f*)(z + o)    = zt;
    *(v2f*)(outb + o) = t;
}

// ---------------------------------------------------------------------------
extern "C" void kernel_launch(void* const* d_in, const int* in_sizes, int n_in,
                              void* d_out, int out_size, void* d_ws, size_t ws_size,
                              hipStream_t stream) {
    const float* x         = (const float*)d_in[0];
    const int*   edge_row  = (const int*)  d_in[1];
    const int*   edge_col  = (const int*)  d_in[2];
    const float* edge_vals = (const float*)d_in[3];
    const float* W1        = (const float*)d_in[4];
    const float* b1        = (const float*)d_in[5];
    const float* W2        = (const float*)d_in[6];
    const float* b2        = (const float*)d_in[7];
    const float* gamma     = (const float*)d_in[8];

    const int IN = 512, HID = 256, C = 64, KCH = 8;
    const int N = in_sizes[0] / IN;
    const int E = in_sizes[1];
    float* zout = (float*)d_out;

    // workspace carve-out (256B aligned slices)
    char* w = (char*)d_ws;
    auto carve = [&](size_t bytes) -> char* {
        char* p = w;
        w += (bytes + 255) & ~(size_t)255;
        return p;
    };
    float*          h       = (float*)carve((size_t)N * C * sizeof(float));
    float*          tA      = (float*)carve((size_t)N * C * sizeof(float));
    float*          tB      = (float*)carve((size_t)N * C * sizeof(float));
    unsigned short* h1      = (unsigned short*)carve((size_t)N * HID * sizeof(unsigned short));
    int*            row_ptr = (int*)carve((size_t)(N + 1) * sizeof(int));
    (void)ws_size; (void)n_in; (void)out_size;

    // 1) CSR row pointers from sorted edge_row
    build_row_ptr_kernel<<<(N + 1 + 255) / 256, 256, 0, stream>>>(edge_row, E, row_ptr, N);

    // 2) h1 = relu(x @ W1 + b1)  -> bf16
    {
        dim3 grid((N + 63) / 64, HID / 64);
        gemm_wmma_kernel<false, true, true><<<grid, 128, 0, stream>>>(x, W1, b1, h1, N, IN, HID);
    }
    // 3) h = h1 @ W2 + b2 -> f32
    {
        dim3 grid((N + 63) / 64, C / 64);
        gemm_wmma_kernel<true, false, false><<<grid, 128, 0, stream>>>(h1, W2, b2, h, N, HID, C);
    }
    // 4) z = gamma0 * h
    {
        int nquads = (N * C) / 4;
        zinit_kernel<<<(nquads + 255) / 256, 256, 0, stream>>>(h, gamma, zout, nquads);
    }

    // 5) t_curr = spmm(h); z += gamma1 * t_curr
    spmm_cheb_kernel<<<N, 32, 0, stream>>>(h, nullptr, tA, zout, row_ptr, edge_col,
                                           edge_vals, gamma, 1, 1.0f, 0);
    // 6) k = 2..8: t_next = 2*spmm(t_curr) - t_prev; z += gamma_k * t_next
    float* prev = h; float* curr = tA; float* nxt = tB;
    for (int k = 2; k <= KCH; ++k) {
        spmm_cheb_kernel<<<N, 32, 0, stream>>>(curr, prev, nxt, zout, row_ptr, edge_col,
                                               edge_vals, gamma, k, 2.0f, 1);
        float* tmp = prev; prev = curr; curr = nxt; nxt = tmp;
    }
}
